// WeightedBatchSubTreeEncoder_84817014161755
// MI455X (gfx1250) — matrix operand order repaired and checked by reading, not compiled, not run on hardware
//
#include <hip/hip_runtime.h>

typedef __bf16 bf16;
typedef __attribute__((ext_vector_type(16))) __bf16 v16bf;
typedef __attribute__((ext_vector_type(8)))  __bf16 v8bf;
typedef __attribute__((ext_vector_type(8)))  float  v8f;

#define LDSW 136   // bf16 halves per LDS row: 16B-aligned, conflict-free b128 loads
#define AGGW 132   // f32 words per agg LDS row: wave halves hit disjoint banks

__global__ void zero_f32_kernel(float* __restrict__ p, int n) {
  int i = blockIdx.x * blockDim.x + threadIdx.x;
  if (i < n) p[i] = 0.0f;
}

// Pack W[c][k] (128x128 f32) into WMMA bf16 B-fragment order:
// flat = ((ct*4 + ks)*32 + lane)*16 + i ; value = W[ct*16 + lane%16][ks*32 + (i/8)*16 + (lane/16)*8 + i%8]
__global__ void pack_w_kernel(const float* __restrict__ Wc, const float* __restrict__ Ws,
                              bf16* __restrict__ pWc, bf16* __restrict__ pWs) {
  int idx = blockIdx.x * blockDim.x + threadIdx.x;       // 0..32767
  const float* src = (idx < 16384) ? Wc : Ws;
  bf16*        dst = (idx < 16384) ? pWc : pWs;
  int t    = idx & 16383;
  int i    = t & 15;
  int lane = (t >> 4) & 31;
  int ks   = (t >> 9) & 3;
  int ct   = t >> 11;
  int c = ct * 16 + (lane & 15);
  int k = ks * 32 + ((i >> 3) << 4) + ((lane >> 4) << 3) + (i & 7);
  dst[t] = (bf16)src[c * 128 + k];
}

// 16x32 bf16 A-fragment from LDS row (ISA layout): element i ->
// K = ks*32 + (i/8)*16 + laneHalf*8 + i%8 ; row pre-offset by laneHalf*8.
__device__ __forceinline__ v16bf load_a_frag(const bf16* row, int ks) {
  v8bf lo = *(const v8bf*)(row + ks * 32);
  v8bf hi = *(const v8bf*)(row + ks * 32 + 16);
  v16bf a;
#pragma unroll
  for (int u = 0; u < 8; ++u) { a[u] = lo[u]; a[u + 8] = hi[u]; }
  return a;
}

__launch_bounds__(128)
__global__ void tree_level_kernel(
    const int*   __restrict__ tokens,   // [B][Ntok]
    const float* __restrict__ emb,      // [V][128]
    const float* __restrict__ Wc_b,     // [128]
    const float* __restrict__ Ws_b,     // [128]
    const bf16*  __restrict__ pWc,      // packed 16384
    const bf16*  __restrict__ pWs,      // packed 16384
    const float* __restrict__ aggIn,    // [chunk][aggInNodes][128] or null
    float*       __restrict__ aggOut,   // [chunk][aggOutNodes][128] or null
    float*       __restrict__ outMax,   // [B][128], nonneg, atomic-int max
    int nodeStart, int aggInNodes, int aggOutNodes,
    int chunkBase, int Ntok, int nodesPerGroup, int doProj)
{
  __shared__ bf16  Elds[16 * LDSW];
  __shared__ bf16  Hlds[16 * LDSW];
  __shared__ float Alds[16 * AGGW];   // staged child-aggregation tile (f32)

  const int tid   = threadIdx.x;
  const int lane  = tid & 31;
  const int wave  = tid >> 5;
  const int laneN = lane & 15;   // B/C/D column, A row (M)
  const int laneH = lane >> 4;   // lane-half select
  const int group = blockIdx.x;  // sibling group == parent node index
  const int btile = blockIdx.y;
  const int bBase = chunkBase + btile * 16;  // global batch base
  const int bLoc  = btile * 16;              // chunk-local batch base
  const int ct0   = wave * 2;                // this wave's two column tiles
  const int hasAgg = (aggIn != nullptr);

  // loop-invariant per-thread biases
  float biasC[2], biasS[2];
#pragma unroll
  for (int c = 0; c < 2; ++c) {
    int ch = (ct0 + c) * 16 + laneN;
    biasC[c] = Wc_b[ch];
    biasS[c] = Ws_b[ch];
  }

  v8f pacc[2];
  float rmax[2][8];
#pragma unroll
  for (int c = 0; c < 2; ++c)
#pragma unroll
    for (int j = 0; j < 8; ++j) { pacc[c][j] = 0.0f; rmax[c][j] = 0.0f; }

  // staging thread roles
  const int sr = tid >> 3;   // 0..15 batch row within tile
  const int sq = tid & 7;    // 16-wide column chunk

  for (int jn = 0; jn < nodesPerGroup; ++jn) {
    const int nodeLocal = group * nodesPerGroup + jn;
    const int col = nodeStart + nodeLocal;

    // --- stage gathered embedding rows (bf16) + child aggregation (f32) ---
    {
      int tok = tokens[(size_t)(bBase + sr) * Ntok + col];
      const float* erow = emb + (size_t)tok * 128 + sq * 16;
      bf16* drow = Elds + sr * LDSW + sq * 16;
#pragma unroll
      for (int u = 0; u < 16; u += 4) {
        float4 v = *(const float4*)(erow + u);
        drow[u + 0] = (bf16)v.x; drow[u + 1] = (bf16)v.y;
        drow[u + 2] = (bf16)v.z; drow[u + 3] = (bf16)v.w;
      }
      if (hasAgg) {
        const float* grow =
            aggIn + ((size_t)(bLoc + sr) * aggInNodes + nodeLocal) * 128 + sq * 16;
        float* arow = Alds + sr * AGGW + sq * 16;
#pragma unroll
        for (int u = 0; u < 16; u += 4)
          *(float4*)(arow + u) = *(const float4*)(grow + u);
      }
      if (jn + 1 < nodesPerGroup) {   // prefetch next sibling's embedding rows
        int tok2 = tokens[(size_t)(bBase + sr) * Ntok + col + 1];
        __builtin_prefetch(emb + (size_t)tok2 * 128 + sq * 16, 0, 3);
      }
    }
    __syncthreads();

    // --- init accumulator with bias (+ staged aggregation) as WMMA C ---
    v8f acc[2];
    if (hasAgg) {
#pragma unroll
      for (int c = 0; c < 2; ++c) {
        int ch = (ct0 + c) * 16 + laneN;
#pragma unroll
        for (int j = 0; j < 8; ++j) {
          int m = j + laneH * 8;
          acc[c][j] = biasC[c] + Alds[m * AGGW + ch];
        }
      }
    } else {
#pragma unroll
      for (int c = 0; c < 2; ++c)
#pragma unroll
        for (int j = 0; j < 8; ++j) acc[c][j] = biasC[c];
    }

    // --- h = E @ Wc^T + C : v_wmma_f32_16x16x32_bf16, K=128 in 4 steps ---
    const bf16* arow = Elds + laneN * LDSW + laneH * 8;
#pragma unroll
    for (int ks = 0; ks < 4; ++ks) {
      v16bf a = load_a_frag(arow, ks);
#pragma unroll
      for (int c = 0; c < 2; ++c) {
        v16bf b = *(const v16bf*)(pWc + (size_t)(((ct0 + c) * 4 + ks) * 32 + lane) * 16);
        acc[c] = __builtin_amdgcn_wmma_f32_16x16x32_bf16(
            false, a, false, b, (short)0, acc[c], false, false);
      }
    }

    // --- post: relu-max, stash h (bf16) for projection ---
#pragma unroll
    for (int c = 0; c < 2; ++c) {
      int ch = (ct0 + c) * 16 + laneN;
#pragma unroll
      for (int j = 0; j < 8; ++j) {
        int m = j + laneH * 8;
        float h = acc[c][j];
        float hr = h > 0.0f ? h : 0.0f;
        rmax[c][j] = fmaxf(rmax[c][j], hr);
        Hlds[m * LDSW + ch] = (bf16)h;
      }
    }
    __syncthreads();

    // --- proj accumulate: pacc += H @ Ws^T (summed over siblings) ---
    if (doProj) {
      const bf16* hrow = Hlds + laneN * LDSW + laneH * 8;
#pragma unroll
      for (int ks = 0; ks < 4; ++ks) {
        v16bf a = load_a_frag(hrow, ks);
#pragma unroll
        for (int c = 0; c < 2; ++c) {
          v16bf b = *(const v16bf*)(pWs + (size_t)(((ct0 + c) * 4 + ks) * 32 + lane) * 16);
          pacc[c] = __builtin_amdgcn_wmma_f32_16x16x32_bf16(
              false, a, false, b, (short)0, pacc[c], false, false);
        }
      }
    }
    __syncthreads();   // Elds/Hlds/Alds reused next sibling
  }

  // --- write parent aggregation tile (exclusive ownership, plain stores) ---
  if (doProj && aggOut) {
#pragma unroll
    for (int c = 0; c < 2; ++c) {
      int ch = (ct0 + c) * 16 + laneN;
      float pb = (float)nodesPerGroup * biasS[c];
#pragma unroll
      for (int j = 0; j < 8; ++j) {
        int m = j + laneH * 8;
        aggOut[((size_t)(bLoc + m) * aggOutNodes + group) * 128 + ch] = pacc[c][j] + pb;
      }
    }
  }

  // --- fold running relu-max into output (nonneg floats: int max preserves order) ---
#pragma unroll
  for (int c = 0; c < 2; ++c) {
    int ch = (ct0 + c) * 16 + laneN;
#pragma unroll
    for (int j = 0; j < 8; ++j) {
      int m = j + laneH * 8;
      atomicMax((int*)&outMax[(size_t)(bBase + m) * 128 + ch], __float_as_int(rmax[c][j]));
    }
  }
}

extern "C" void kernel_launch(void* const* d_in, const int* in_sizes, int n_in,
                              void* d_out, int out_size, void* d_ws, size_t ws_size,
                              hipStream_t stream) {
  (void)in_sizes; (void)n_in;
  const int*   tokens = (const int*)d_in[0];
  const float* emb    = (const float*)d_in[1];
  const float* WcW    = (const float*)d_in[2];
  const float* Wcb    = (const float*)d_in[3];
  const float* WsW    = (const float*)d_in[4];
  const float* Wsb    = (const float*)d_in[5];
  float* out = (float*)d_out;

  const int B = 512, Ntok = 1365;

  // workspace: packed bf16 weights (2 x 32KB) + ping/pong agg buffers
  char* ws  = (char*)d_ws;
  bf16* pWc = (bf16*)ws;
  bf16* pWs = pWc + 16384;
  float* ping = (float*)(ws + 65536);

  // largest batch chunk that fits: need = 64KB + chunk*(256+64)*128*4 bytes
  int chunk = 512;
  while (chunk > 16 && (size_t)65536 + (size_t)chunk * 163840u > ws_size) chunk >>= 1;
  float* pong = ping + (size_t)chunk * 256 * 128;

  zero_f32_kernel<<<(out_size + 255) / 256, 256, 0, stream>>>(out, out_size);
  pack_w_kernel<<<32768 / 256, 256, 0, stream>>>(WcW, WsW, pWc, pWs);

  static const int startsL[6] = {0, 1, 5, 21, 85, 341};
  static const int sizesL[6]  = {1, 4, 16, 64, 256, 1024};
  // agg buffers ping-pong down the tree; indexed by level d
  float* aggInP[6]  = {ping,    pong, ping, pong, ping, nullptr};
  float* aggOutP[6] = {nullptr, ping, pong, ping, pong, ping};

  for (int c0 = 0; c0 < B; c0 += chunk) {
    int btiles = chunk / 16;
    for (int d = 5; d >= 0; --d) {
      int npg = (d == 0) ? 1 : 4;
      dim3 grid(sizesL[d] / npg, btiles);
      tree_level_kernel<<<grid, 128, 0, stream>>>(
          tokens, emb, Wcb, Wsb, pWc, pWs,
          aggInP[d], aggOutP[d], out,
          startsL[d], sizesL[d], (d > 0) ? sizesL[d - 1] : 0,
          c0, Ntok, npg, (d > 0) ? 1 : 0);
    }
  }
}
